// Attention_31619549233801
// MI455X (gfx1250) — matrix operand order repaired and checked by reading, not compiled, not run on hardware
//
#include <hip/hip_runtime.h>
#include <hip/hip_bf16.h>

// ---------------------------------------------------------------------------
// Problem constants (from the reference)
// ---------------------------------------------------------------------------
constexpr int BATCH = 4;
constexpr int NN    = 2048;
constexpr int IND   = 256;   // IN_DIM
constexpr int HH    = 8;     // heads
constexpr int AT    = 32;    // ATN = OUT_DIM / H
constexpr int OUTD  = HH * AT;  // 256

// ---------------------------------------------------------------------------
// Vector types for WMMA fragments (wave32, 16x16x32 bf16 -> f32)
// ---------------------------------------------------------------------------
typedef __attribute__((ext_vector_type(16))) __bf16 v16bf;
typedef __attribute__((ext_vector_type(8)))  __bf16 bf16x8;
typedef __attribute__((ext_vector_type(8)))  float  v8f;

static __device__ __forceinline__ v16bf mk16(bf16x8 lo, bf16x8 hi) {
  v16bf r;
#pragma unroll
  for (int i = 0; i < 8; ++i) { r[i] = lo[i]; r[8 + i] = hi[i]; }
  return r;
}

static __device__ __forceinline__ v8f wmma_bf16(v16bf a, v16bf b, v8f c) {
  // (neg_a, A, neg_b, B, c_mod, C, reuse_a, reuse_b)
  return __builtin_amdgcn_wmma_f32_16x16x32_bf16(false, a, false, b,
                                                 (short)0, c, false, false);
}

// Hardware transcendental tanh (CDNA5 V_TANH_F32, TRANS32 pipe).
static __device__ __forceinline__ float fast_tanh(float x) {
#if __has_builtin(__builtin_amdgcn_tanhf)
  return __builtin_amdgcn_tanhf(x);
#else
  float r;
  // v_nop: TRANS ops need 1 independent op before the result is consumed;
  // the compiler cannot see through the asm to schedule one.
  asm volatile("v_tanh_f32 %0, %1\n\tv_nop" : "=v"(r) : "v"(x));
  return r;
#endif
}

// ---------------------------------------------------------------------------
// Kernel 1: convert weights.  Wtb[h][o][i] = bf16(W[h][i][o]) ;
//           Cb[h][p][o]      = bf16(C[h][p][o])
// ---------------------------------------------------------------------------
__global__ __launch_bounds__(256) void prep_kernel(
    const float* __restrict__ W, const float* __restrict__ C,
    __bf16* __restrict__ Wtb, __bf16* __restrict__ Cb) {
  int i = blockIdx.x * blockDim.x + threadIdx.x;
  if (i < HH * AT * IND) {
    int h  = i / (AT * IND);
    int o  = (i / IND) % AT;
    int ii = i % IND;
    Wtb[i] = (__bf16)W[((size_t)h * IND + ii) * AT + o];
  }
  if (i < HH * AT * AT) {
    Cb[i] = (__bf16)C[i];
  }
}

// ---------------------------------------------------------------------------
// Kernel 2: projection.  One wave per (b,h,16-row tile of N).
//   xt  = x @ W[h] + bias[h]          (K = 256, 8 WMMA chunks x 2 o-tiles)
//   xC  = xt @ C[h]^T                 (K = 32, 1 chunk x 2 p-tiles)
// Stores xt [n][32], xtT [32][n], xC [n][32] as bf16.
// ---------------------------------------------------------------------------
__global__ __launch_bounds__(256) void proj_kernel(
    const float* __restrict__ x, const float* __restrict__ bias,
    const __bf16* __restrict__ Wtb, const __bf16* __restrict__ Cb,
    __bf16* __restrict__ xt, __bf16* __restrict__ xtT,
    __bf16* __restrict__ xC) {
  __shared__ __bf16 smem[8][16 * AT];

  const int lane = threadIdx.x & 31;
  const int wid  = threadIdx.x >> 5;
  const int flat = blockIdx.x * 8 + wid;           // 4096 waves total
  const int mt   = flat & 127;                     // N/16 tiles
  const int h    = (flat >> 7) & 7;
  const int bb   = flat >> 10;
  const int nbase = mt * 16;
  const int l15  = lane & 15;
  const int half = lane >> 4;

  const float*  xrow = x + ((size_t)bb * NN + (nbase + l15)) * IND;
  const __bf16* wt0  = Wtb + ((size_t)h * AT + l15) * IND;
  const __bf16* wt1  = Wtb + ((size_t)h * AT + 16 + l15) * IND;

  v8f acc0, acc1;
  const float bv0 = bias[h * AT + l15];
  const float bv1 = bias[h * AT + 16 + l15];
#pragma unroll
  for (int i = 0; i < 8; ++i) { acc0[i] = bv0; acc1[i] = bv1; }

  const int kbA = half ? 16 : 0;   // A-fragment K base (one run of 16)
  const int kbB = half ? 16 : 0;   // B-fragment K base (one run of 16)

#pragma unroll
  for (int ki = 0; ki < IND; ki += 32) {
    v16bf a;
#pragma unroll
    for (int i = 0; i < 8; ++i) a[i]     = (__bf16)xrow[ki + kbA + i];
#pragma unroll
    for (int i = 0; i < 8; ++i) a[8 + i] = (__bf16)xrow[ki + kbA + 8 + i];
    v16bf b0 = *(const v16bf*)(wt0 + ki + kbB);
    v16bf b1 = *(const v16bf*)(wt1 + ki + kbB);
    acc0 = wmma_bf16(a, b0, acc0);
    acc1 = wmma_bf16(a, b1, acc1);
  }

  __bf16* xt_bh  = xt  + ((size_t)bb * HH + h) * NN * AT;
  __bf16* xtT_bh = xtT + ((size_t)bb * HH + h) * AT * NN;
#pragma unroll
  for (int r = 0; r < 8; ++r) {
    const int M   = half * 8 + r;           // C-layout row
    const int row = nbase + M;
    __bf16 e0 = (__bf16)acc0[r];
    __bf16 e1 = (__bf16)acc1[r];
    xt_bh[(size_t)row * AT + l15]        = e0;
    xt_bh[(size_t)row * AT + 16 + l15]   = e1;
    xtT_bh[(size_t)l15 * NN + row]       = e0;
    xtT_bh[(size_t)(16 + l15) * NN + row] = e1;
    smem[wid][M * AT + l15]      = e0;
    smem[wid][M * AT + 16 + l15] = e1;
  }
  __syncthreads();

  // xC = xt @ C^T : A fragment re-read from LDS in A-layout
  const __bf16* sp = &smem[wid][l15 * AT];
  v16bf a2 = mk16(*(const bf16x8*)(sp + kbA), *(const bf16x8*)(sp + kbA + 8));
  v16bf c0 = *(const v16bf*)(Cb + ((size_t)h * AT + l15) * AT + kbB);
  v16bf c1 = *(const v16bf*)(Cb + ((size_t)h * AT + 16 + l15) * AT + kbB);
  v8f p0 = {}, p1 = {};
  p0 = wmma_bf16(a2, c0, p0);
  p1 = wmma_bf16(a2, c1, p1);

  __bf16* xC_bh = xC + ((size_t)bb * HH + h) * NN * AT;
#pragma unroll
  for (int r = 0; r < 8; ++r) {
    const int row = nbase + half * 8 + r;
    xC_bh[(size_t)row * AT + l15]      = (__bf16)p0[r];
    xC_bh[(size_t)row * AT + 16 + l15] = (__bf16)p1[r];
  }
}

// ---------------------------------------------------------------------------
// Kernel 3: fused scores -> tanh(adj*s) -> heads.  One wave per
// (b,h,16-row tile).  Loops over N in chunks of 32:
//   S(16x32)   : 2x wmma  (A = xC rows, B = xt rows, K = o = 32)
//   alpha      : adj-mask + v_tanh_f32 in C-layout, re-layout via LDS tile
//   acc(16x32) : 2x wmma  (A = alpha, B = xtT, K = n chunk = 32)
// adj tile is software-pipelined one chunk ahead (registers) and prefetched
// two chunks ahead (global_prefetch).
// ---------------------------------------------------------------------------
__global__ __launch_bounds__(256) void attn_kernel(
    const float* __restrict__ adj, const __bf16* __restrict__ xt,
    const __bf16* __restrict__ xtT, const __bf16* __restrict__ xC,
    float* __restrict__ out) {
  __shared__ __bf16 smem[8][16 * AT];

  const int lane = threadIdx.x & 31;
  const int wid  = threadIdx.x >> 5;
  const int flat = blockIdx.x * 8 + wid;
  const int mt   = flat & 127;
  const int h    = (flat >> 7) & 7;
  const int bb   = flat >> 10;
  const int mbase = mt * 16;
  const int l15  = lane & 15;
  const int half = lane >> 4;
  const int kbA  = half ? 16 : 0;
  const int kbB  = half ? 16 : 0;

  const __bf16* xt_bh  = xt  + ((size_t)bb * HH + h) * NN * AT;
  const __bf16* xtT_bh = xtT + ((size_t)bb * HH + h) * AT * NN;
  const __bf16* xC_bh  = xC  + ((size_t)bb * HH + h) * NN * AT;
  const float*  adj_b  = adj + (size_t)bb * NN * NN;

  // A fragment: xC rows mbase..mbase+15, K = p = 0..31 (loaded once, reused)
  const __bf16* ap = xC_bh + (size_t)(mbase + l15) * AT;
  const v16bf a = mk16(*(const bf16x8*)(ap + kbA), *(const bf16x8*)(ap + kbA + 8));

  v8f acc0 = {}, acc1 = {};

  // Software-pipelined adj tile registers (C-layout: this lane's 16 elements)
  float ad0[8], ad1[8];
#pragma unroll
  for (int r = 0; r < 8; ++r) {
    const float* p = adj_b + (size_t)(mbase + half * 8 + r) * NN;
    ad0[r] = p[l15];
    ad1[r] = p[16 + l15];
  }

  for (int nb = 0; nb < NN; nb += 32) {
    // scores: B tiles = xt rows (K = o contiguous in xt[n][32])
    v16bf sb0 = *(const v16bf*)(xt_bh + (size_t)(nb + l15) * AT + kbB);
    v16bf sb1 = *(const v16bf*)(xt_bh + (size_t)(nb + 16 + l15) * AT + kbB);
    v8f s0 = {}, s1 = {};
    s0 = wmma_bf16(a, sb0, s0);
    s1 = wmma_bf16(a, sb1, s1);

    // prefetch adj for chunk nb+32 into registers (covered by this chunk's
    // WMMA latency), and prime caches for nb+64
    float nx0[8], nx1[8];
    if (nb + 32 < NN) {
#pragma unroll
      for (int r = 0; r < 8; ++r) {
        const float* p = adj_b + (size_t)(mbase + half * 8 + r) * NN + nb + 32;
        nx0[r] = p[l15];
        nx1[r] = p[16 + l15];
      }
      if (nb + 64 < NN)
        __builtin_prefetch(adj_b + (size_t)(mbase + half * 8) * NN + nb + 64 + l15, 0, 3);
    }

    __syncthreads();  // WAR vs. previous iteration's LDS reads
#pragma unroll
    for (int r = 0; r < 8; ++r) {
      const int M = half * 8 + r;
      smem[wid][M * AT + l15]      = (__bf16)fast_tanh(s0[r] * ad0[r]);
      smem[wid][M * AT + 16 + l15] = (__bf16)fast_tanh(s1[r] * ad1[r]);
    }
    __syncthreads();

    // alpha re-layout: C-format tile in LDS -> A-format fragment
    const __bf16* sp = &smem[wid][l15 * AT];
    v16bf af = mk16(*(const bf16x8*)(sp + kbA), *(const bf16x8*)(sp + kbA + 8));

    // heads: B tiles from xtT (K = n contiguous)
    v16bf hb0 = *(const v16bf*)(xtT_bh + (size_t)l15 * NN + nb + kbB);
    v16bf hb1 = *(const v16bf*)(xtT_bh + (size_t)(16 + l15) * NN + nb + kbB);
    acc0 = wmma_bf16(af, hb0, acc0);
    acc1 = wmma_bf16(af, hb1, acc1);

#pragma unroll
    for (int r = 0; r < 8; ++r) { ad0[r] = nx0[r]; ad1[r] = nx1[r]; }
  }

  // out[b][n][h*32 + o], float32
#pragma unroll
  for (int r = 0; r < 8; ++r) {
    const int row = mbase + half * 8 + r;
    float* op = out + ((size_t)bb * NN + row) * OUTD + h * AT;
    op[l15]      = acc0[r];
    op[16 + l15] = acc1[r];
  }
}

// ---------------------------------------------------------------------------
// Host launcher
// ---------------------------------------------------------------------------
extern "C" void kernel_launch(void* const* d_in, const int* in_sizes, int n_in,
                              void* d_out, int out_size, void* d_ws,
                              size_t ws_size, hipStream_t stream) {
  const float* x    = (const float*)d_in[0];
  const float* adj  = (const float*)d_in[1];
  const float* W    = (const float*)d_in[2];
  const float* bias = (const float*)d_in[3];
  const float* C    = (const float*)d_in[4];

  char* ws = (char*)d_ws;
  const size_t SZ = (size_t)BATCH * HH * NN * AT * sizeof(__bf16);  // 4 MiB
  __bf16* xt  = (__bf16*)(ws);
  __bf16* xtT = (__bf16*)(ws + SZ);
  __bf16* xC  = (__bf16*)(ws + 2 * SZ);
  __bf16* Wtb = (__bf16*)(ws + 3 * SZ);
  __bf16* Cb  = (__bf16*)(ws + 3 * SZ + 256 * 1024);

  // 1) weight conversion / transpose
  prep_kernel<<<(HH * AT * IND + 255) / 256, 256, 0, stream>>>(W, C, Wtb, Cb);

  // 2) projection + bilinear-left (xC); 4096 waves = 512 blocks x 8 waves
  proj_kernel<<<512, 256, 0, stream>>>(x, bias, Wtb, Cb, xt, xtT, xC);

  // 3) fused scores/tanh/heads
  attn_kernel<<<512, 256, 0, stream>>>(adj, xt, xtT, xC, (float*)d_out);
}
